// relaxRNN_20246475833379
// MI455X (gfx1250) — matrix operand order repaired and checked by reading, not compiled, not run on hardware
//
#include <hip/hip_runtime.h>
#include <hip/hip_bf16.h>

// ---------------- problem constants (from reference) ----------------
#define BATCH    256
#define TSTEPS   1024
#define IN_DIM   96
#define NUNITS   256
#define NCLS     10
#define EPS_C    0.01f
#define BETA_C   0.5f
#define GAMMA_C  0.0f

// ---------------- WMMA types ----------------
typedef __attribute__((ext_vector_type(16))) __bf16 v16bf;
typedef __attribute__((ext_vector_type(8)))  float  v8f;

union Frag {            // one 16x32 (A) or 32x16 (B) bf16 WMMA operand: 8 dwords/lane
    uint4    q[2];
    unsigned u[8];
    v16bf    v;
};

// float -> bf16 (round half up: bias then truncate). One v_add + use of high half.
static __device__ __forceinline__ unsigned short f2bf(float f) {
    return (unsigned short)((__float_as_uint(f) + 0x8000u) >> 16);
}
// pack two floats to bf16x2: 2 bias adds + 1 v_perm_b32
static __device__ __forceinline__ unsigned pk2bf(float a, float b) {
    unsigned ua = __float_as_uint(a) + 0x8000u;
    unsigned ub = __float_as_uint(b) + 0x8000u;
    // result bytes: [1:0] = ua[3:2], [3:2] = ub[3:2]  (sel 0..3 -> s1, 4..7 -> s0)
    return __builtin_amdgcn_perm(ub, ua, 0x07060302u);
}

static __device__ __forceinline__ float fast_tanh(float v) {
#if defined(__has_builtin)
#if __has_builtin(__builtin_amdgcn_tanhf)
    return __builtin_amdgcn_tanhf(v);   // v_tanh_f32 on gfx1250
#else
    return tanhf(v);
#endif
#else
    return tanhf(v);
#endif
}

// A[k][n] = beta*(C[k,n]-C[n,k]) + (1-beta)*(C[k,n]+C[n,k]) - gamma*(k==n)
static __device__ __forceinline__ float mkA(const float* M, int k, int n) {
    float a = M[k * NUNITS + n];
    float b = M[n * NUNITS + k];
    float v = BETA_C * (a - b) + (1.0f - BETA_C) * (a + b);
    if (k == n) v -= GAMMA_C;
    return v;
}

static __device__ __forceinline__ v8f wmma_bf16(const Frag& a, const Frag& b, v8f c) {
    return __builtin_amdgcn_wmma_f32_16x16x32_bf16(false, a.v, false, b.v,
                                                   (short)0, c, false, false);
}

// 16 workgroups x 512 threads (16 wave32). Block bi owns batch rows
// [16*bi, 16*bi+16). Wave w owns unit columns [16*w, 16*w+16).
__global__ void __launch_bounds__(512, 1)
relax_rnn_fused(const float* __restrict__ x,     // [B,T,F]
                const float* __restrict__ Ew,    // [N,F]
                const float* __restrict__ Eb,    // [N]
                const float* __restrict__ Cm,    // [N,N]
                const float* __restrict__ Bm,    // [N,N]
                const float* __restrict__ Dw,    // [CLS,N]
                const float* __restrict__ Db,    // [CLS]
                float* __restrict__ out)         // [B,CLS]
{
    __shared__ __align__(16) unsigned short hbf[2][16][NUNITS]; // bf16 h, double buffered
    __shared__ __align__(16) unsigned short xbf[2][16][IN_DIM]; // bf16 x_t, double buffered
    __shared__ __align__(16) float hfin[16][NUNITS];            // fp32 h for epilogue

    const int tid   = threadIdx.x;
    const int wave  = tid >> 5;        // 0..15  -> owned N-tile
    const int lane  = tid & 31;
    const int half  = lane >> 4;       // 0: lanes 0-15, 1: lanes 16-31
    const int l16   = lane & 15;
    const int kbA   = half ? 16 : 0;   // A-matrix per-lane K window base (contiguous, matches B)
    const int ncol  = wave * 16 + l16; // global unit column (B/C/D layout N = lane%16)
    const int b0    = blockIdx.x * 16;
    const int mbase = half ? 8 : 0;    // C/D layout: VGPR r -> M = r + mbase

    // cooperative x stager: 384 threads each own one float4 of the 16x96 tile
    const int  sm  = tid / 24;              // row within tile
    const int  sf  = (tid % 24) * 4;        // float offset within row
    const bool sact = (tid < 384);
    const float* xsrc = x + (((size_t)(b0 + sm)) * TSTEPS) * IN_DIM + sf;

    // ---- zero-init h buffers ----
    for (int i = tid; i < 2 * 16 * NUNITS; i += 512)
        ((unsigned short*)hbf)[i] = 0;

    // ---- build resident bf16 B-fragments for A, Dm, E_w (one time) ----
    // B-matrix 32x16 in this code's contiguous k order (same as A loads): lanes 0-15 K=0..15,
    // lanes 16-31 K=16..31; VGPR v holds K = 16*half + 2v, 2v+1.
    Frag Af[8], Df[8], Ef[3];
#pragma unroll
    for (int kk = 0; kk < 8; ++kk) {
#pragma unroll
        for (int v = 0; v < 8; ++v) {
            int k = 32 * kk + 16 * half + 2 * v;
            Af[kk].u[v] = pk2bf(mkA(Cm, k, ncol), mkA(Cm, k + 1, ncol));
            Df[kk].u[v] = pk2bf(mkA(Bm, k, ncol), mkA(Bm, k + 1, ncol));
        }
    }
#pragma unroll
    for (int kk = 0; kk < 3; ++kk) {
#pragma unroll
        for (int v = 0; v < 8; ++v) {
            int f = 32 * kk + 16 * half + 2 * v;       // < 96
            Ef[kk].u[v] = pk2bf(Ew[ncol * IN_DIM + f], Ew[ncol * IN_DIM + f + 1]);
        }
    }
    const float eb = Eb[ncol];

    float hf[8];                       // fp32 master h, C-layout (this wave's N-tile)
#pragma unroll
    for (int r = 0; r < 8; ++r) hf[r] = 0.0f;

    // ---- preload x for t = 0 ----
    float4 xstage;
    if (sact) xstage = *(const float4*)xsrc;
    if (sact) {
        unsigned lo = pk2bf(xstage.x, xstage.y);
        unsigned hi = pk2bf(xstage.z, xstage.w);
        *(uint2*)&xbf[0][sm][sf] = make_uint2(lo, hi);
    }
    __syncthreads();

    // =================== sequential recurrence over T ===================
    for (int t = 0; t < TSTEPS; ++t) {
        const int p = t & 1;

        // issue next step's x loads early; latency hidden by this step's WMMAs
        const bool more = (t + 1 < TSTEPS);
        if (sact && more)
            xstage = *(const float4*)(xsrc + (size_t)(t + 1) * IN_DIM);

        v8f acc1 = {0.f,0.f,0.f,0.f,0.f,0.f,0.f,0.f};   // -> z + h@A
        v8f acc2 = {0.f,0.f,0.f,0.f,0.f,0.f,0.f,0.f};   // -> h@Dm

        // ---- fused input projection: acc1 = x_t @ E_w^T  (3 WMMAs) ----
        const unsigned short* xr = &xbf[p][l16][0];
#pragma unroll
        for (int kk = 0; kk < 3; ++kk) {
            const int f0 = 32 * kk + kbA;              // A-layout K window base
            Frag xf;
            xf.q[0] = *(const uint4*)(xr + f0);        // ds_load_b128
            xf.q[1] = *(const uint4*)(xr + f0 + 8);    // ds_load_b128
            acc1 = wmma_bf16(xf, Ef[kk], acc1);
        }

        // ---- recurrent GEMMs: acc1 += h@A ; acc2 += h@Dm (16 WMMAs) ----
        const unsigned short* hr = &hbf[p][l16][0];
#pragma unroll
        for (int kk = 0; kk < 8; ++kk) {
            const int u0 = 32 * kk + kbA;
            Frag hfrg;
            hfrg.q[0] = *(const uint4*)(hr + u0);        // ds_load_b128
            hfrg.q[1] = *(const uint4*)(hr + u0 + 8);    // ds_load_b128
            acc1 = wmma_bf16(hfrg, Af[kk], acc1);
            acc2 = wmma_bf16(hfrg, Df[kk], acc2);
        }

        // ---- elementwise update + publish bf16 h for next step ----
#pragma unroll
        for (int r = 0; r < 8; ++r) {
            float hn = hf[r] + EPS_C * acc2[r] + EPS_C * fast_tanh(acc1[r] + eb);
            hf[r] = hn;
            hbf[p ^ 1][mbase + r][ncol] = f2bf(hn);      // v_add + ds_store_b16
        }

        // ---- commit next step's x tile to LDS (waits on the early loads) ----
        if (sact && more) {
            unsigned lo = pk2bf(xstage.x, xstage.y);
            unsigned hi = pk2bf(xstage.z, xstage.w);
            *(uint2*)&xbf[p ^ 1][sm][sf] = make_uint2(lo, hi);
        }

        __syncthreads();   // one barrier per step (double-buffered h and x)
    }

    // =================== epilogue: out = h @ D_w^T + D_b ===================
#pragma unroll
    for (int r = 0; r < 8; ++r)
        hfin[mbase + r][ncol] = hf[r];
    __syncthreads();

    if (tid < 16 * NCLS) {
        const int m = tid / NCLS, c = tid % NCLS;
        float s = Db[c];
        const float* dwr = Dw + c * NUNITS;
#pragma unroll 8
        for (int n = 0; n < NUNITS; ++n)
            s += hfin[m][n] * dwr[n];
        out[(size_t)(b0 + m) * NCLS + c] = s;
    }
}

extern "C" void kernel_launch(void* const* d_in, const int* in_sizes, int n_in,
                              void* d_out, int out_size, void* d_ws, size_t ws_size,
                              hipStream_t stream) {
    (void)in_sizes; (void)n_in; (void)out_size; (void)d_ws; (void)ws_size;
    const float* x  = (const float*)d_in[0];
    const float* Ew = (const float*)d_in[1];
    const float* Eb = (const float*)d_in[2];
    const float* Cm = (const float*)d_in[3];
    const float* Bm = (const float*)d_in[4];
    const float* Dw = (const float*)d_in[5];
    const float* Db = (const float*)d_in[6];
    float* out = (float*)d_out;

    relax_rnn_fused<<<dim3(BATCH / 16), dim3(512), 0, stream>>>(
        x, Ew, Eb, Cm, Bm, Dw, Db, out);
}
